// LogWhispererBrain_18262200942710
// MI455X (gfx1250) — matrix operand order repaired and checked by reading, not compiled, not run on hardware
//
#include <hip/hip_runtime.h>

#define N_NODES 50000
#define N_EDGES 800000
#define IN_CH   128
#define HID     64
#define HEADS   4
#define OUT_CH  32

typedef __attribute__((ext_vector_type(16))) __bf16       v16bf;
typedef __attribute__((ext_vector_type(8)))  float        v8f;
typedef __attribute__((ext_vector_type(8)))  unsigned int v8u;

__device__ __forceinline__ unsigned short f32_to_bf16_rne(float f) {
  unsigned u = __float_as_uint(f);
  u = u + 0x7FFFu + ((u >> 16) & 1u);
  return (unsigned short)(u >> 16);
}

// order-preserving float -> uint encoding for atomicMax
__device__ __forceinline__ unsigned enc_f32(float f) {
  unsigned u = __float_as_uint(f);
  return (u & 0x80000000u) ? ~u : (u | 0x80000000u);
}
__device__ __forceinline__ float dec_f32(unsigned u) {
  unsigned b = (u & 0x80000000u) ? (u ^ 0x80000000u) : ~u;
  return __uint_as_float(b);
}

__global__ void k_f32_to_bf16(const float* __restrict__ in,
                              unsigned short* __restrict__ out, int n) {
  int i = blockIdx.x * blockDim.x + threadIdx.x;
  if (i < n) out[i] = f32_to_bf16_rne(in[i]);
}

__global__ void k_elu_to_bf16(const float* __restrict__ in,
                              unsigned short* __restrict__ out, int n) {
  int i = blockIdx.x * blockDim.x + threadIdx.x;
  if (i >= n) return;
  float v = in[i];
  v = (v > 0.f) ? v : (__expf(v) - 1.f);
  out[i] = f32_to_bf16_rne(v);
}

__global__ void k_fill_bias(float* __restrict__ out,
                            const float* __restrict__ bias, int n, int C) {
  int i = blockIdx.x * blockDim.x + threadIdx.x;
  if (i < n) out[i] = bias[i % C];
}

// Pack f32 B[K,N] (row-major) into WMMA bf16 B-operand register layout:
//   Bp[tile_n][kstep][lane][8 dwords]  (each dword = 2 bf16, K and K+1)
//   lane<16: col = tile_n*16+lane, K = k0+0..15 ; lane>=16: K = k0+16..31
__global__ void k_pack_b_bf16(const float* __restrict__ B, unsigned* __restrict__ Bp,
                              int N, int K) {
  const int ksteps = K >> 5;
  int i = blockIdx.x * blockDim.x + threadIdx.x;
  int total = (N >> 4) * ksteps * 32;
  if (i >= total) return;
  int lane   = i & 31;
  int t      = i >> 5;
  int kstep  = t % ksteps;
  int tile_n = t / ksteps;
  int half = lane >> 4, n = lane & 15;
  int col = tile_n * 16 + n;
  int k0  = kstep * 32 + 16 * half;
  unsigned* dst = Bp + (size_t)i * 8;
#pragma unroll
  for (int r = 0; r < 8; ++r) {
    unsigned lo = f32_to_bf16_rne(B[(size_t)(k0 + 2 * r) * N + col]);
    unsigned hi = f32_to_bf16_rne(B[(size_t)(k0 + 2 * r + 1) * N + col]);
    dst[r] = lo | (hi << 16);
  }
}

// C[M,N](f32) = A[M,K](bf16 row-major) * B(pre-packed) ; one wave32 per 16x16 tile.
// A uses contiguous k order (k = 16*half + s, as in packed B), two 16B chunks per lane:
//   half=0: dwords 0..3 (K0..7) + 4..7 (K8..15); half=1: 8..11 + 12..15.
template <int KSTEPS>
__global__ void k_gemm_bf16_wmma(const unsigned short* __restrict__ A,
                                 const unsigned* __restrict__ Bp,
                                 float* __restrict__ C, int M, int N) {
  const int K      = KSTEPS * 32;
  const int lane   = threadIdx.x & 31;
  const int warp   = threadIdx.x >> 5;
  const int tile_m = blockIdx.x;
  const int tile_n = blockIdx.y * (blockDim.x >> 5) + warp;
  const int half   = lane >> 4;
  const int mn     = lane & 15;
  const int row    = tile_m * 16;
  const int col    = tile_n * 16;

  const unsigned short* Arow  = A + (size_t)(row + mn) * K;
  const uint4*          Bbase = (const uint4*)(Bp + ((size_t)tile_n * KSTEPS * 32 + lane) * 8);

  v8f acc = {};
#pragma unroll
  for (int ks = 0; ks < KSTEPS; ++ks) {
    const unsigned short* Ap = Arow + ks * 32;
    uint4 al = *(const uint4*)(Ap + 16 * half);
    uint4 ah = *(const uint4*)(Ap + 16 * half + 8);
    uint4 bl = Bbase[(size_t)ks * 64 + 0];   // 32 lanes * 2 uint4 per kstep
    uint4 bh = Bbase[(size_t)ks * 64 + 1];
    v8u au = {al.x, al.y, al.z, al.w, ah.x, ah.y, ah.z, ah.w};
    v8u bu = {bl.x, bl.y, bl.z, bl.w, bh.x, bh.y, bh.z, bh.w};
    acc = __builtin_amdgcn_wmma_f32_16x16x32_bf16(false, __builtin_bit_cast(v16bf, au),
                                                  false, __builtin_bit_cast(v16bf, bu),
                                                  (short)0, acc, false, false);
  }
#pragma unroll
  for (int i = 0; i < 8; ++i)
    C[(size_t)(row + i + 8 * half) * N + col + mn] = acc[i];
}

// per-(node,head): attention dots; also init softmax state (max enc(-inf), sum=0)
__global__ void k_node_att(const float* __restrict__ h,
                           const float* __restrict__ att_src,
                           const float* __restrict__ att_dst,
                           float* __restrict__ a_src, float* __restrict__ a_dst,
                           unsigned* __restrict__ nmax, float* __restrict__ nsum,
                           int N, int H, int C) {
  int i = blockIdx.x * blockDim.x + threadIdx.x;
  if (i >= N * H) return;
  int n = i / H, hh = i - n * H;
  const float* hp = h + (size_t)n * H * C + hh * C;
  const float* as = att_src + hh * C;
  const float* ad = att_dst + hh * C;
  float s = 0.f, d = 0.f;
  for (int c = 0; c < C; ++c) { float v = hp[c]; s += v * as[c]; d += v * ad[c]; }
  a_src[i] = s; a_dst[i] = d;
  nmax[i] = 0u;   // 0 encodes below -FLT_MAX in enc_f32 ordering
  nsum[i] = 0.f;
}

template <int H>
__global__ void k_edge_logits(const int* __restrict__ src, const int* __restrict__ dst,
                              const float* __restrict__ a_src, const float* __restrict__ a_dst,
                              float* __restrict__ elog, unsigned* __restrict__ nmax, int E) {
  int e = blockIdx.x * blockDim.x + threadIdx.x;
  if (e >= E) return;
  int s = src[e], d = dst[e];
#pragma unroll
  for (int h = 0; h < H; ++h) {
    float l = a_src[s * H + h] + a_dst[d * H + h];
    l = (l > 0.f) ? l : 0.2f * l;                 // leaky_relu
    elog[(size_t)e * H + h] = l;
    atomicMax(&nmax[d * H + h], enc_f32(l));
  }
}

template <int H>
__global__ void k_edge_exp(const int* __restrict__ dst, float* __restrict__ elog,
                           const unsigned* __restrict__ nmax, float* __restrict__ nsum, int E) {
  int e = blockIdx.x * blockDim.x + threadIdx.x;
  if (e >= E) return;
  int d = dst[e];
#pragma unroll
  for (int h = 0; h < H; ++h) {
    float m  = dec_f32(nmax[d * H + h]);
    float ex = __expf(elog[(size_t)e * H + h] - m);
    elog[(size_t)e * H + h] = ex;
    atomicAdd(&nsum[d * H + h], ex);
  }
}

// (H*C/4) threads per edge, each handles one float4 channel chunk
template <int H, int C>
__global__ void k_edge_scatter(const int* __restrict__ src, const int* __restrict__ dst,
                               const float* __restrict__ h, const float* __restrict__ elog,
                               const float* __restrict__ nsum, float* __restrict__ out, int E) {
  constexpr int TPE = (H * C) >> 2;
  long long gid = (long long)blockIdx.x * blockDim.x + threadIdx.x;
  int e = (int)(gid / TPE);
  int t = (int)(gid % TPE);
  if (e >= E) return;
  int s = src[e], d = dst[e];
  int c4 = t * 4;
  int hh = c4 / C;
  float alpha = elog[(size_t)e * H + hh] / (nsum[d * H + hh] + 1e-16f);
  const float4 hv = *(const float4*)(h + (size_t)s * H * C + c4);
  float* op = out + (size_t)d * H * C + c4;
  atomicAdd(op + 0, hv.x * alpha);
  atomicAdd(op + 1, hv.y * alpha);
  atomicAdd(op + 2, hv.z * alpha);
  atomicAdd(op + 3, hv.w * alpha);
}

extern "C" void kernel_launch(void* const* d_in, const int* in_sizes, int n_in,
                              void* d_out, int out_size, void* d_ws, size_t ws_size,
                              hipStream_t stream) {
  const float* x   = (const float*)d_in[0];
  const int*   ei  = (const int*)d_in[1];
  const float* W1  = (const float*)d_in[2];
  const float* as1 = (const float*)d_in[3];
  const float* ad1 = (const float*)d_in[4];
  const float* b1  = (const float*)d_in[5];
  const float* W2  = (const float*)d_in[6];
  const float* as2 = (const float*)d_in[7];
  const float* ad2 = (const float*)d_in[8];
  const float* b2  = (const float*)d_in[9];
  float* out = (float*)d_out;

  const int N = N_NODES, E = N_EDGES, HC1 = HEADS * HID; // 256
  const int* src  = ei;
  const int* dstv = ei + E;

  char* ws = (char*)d_ws;
  size_t off = 0;
  auto carve = [&](size_t bytes) -> void* {
    void* p = ws + off;
    off = (off + bytes + 255) & ~(size_t)255;
    return p;
  };
  unsigned short* x_bf    = (unsigned short*)carve((size_t)N * IN_CH * 2);
  unsigned*       w1_pk   = (unsigned*)carve((size_t)(HC1 / 16) * (IN_CH / 32) * 32 * 8 * 4);
  unsigned*       w2_pk   = (unsigned*)carve((size_t)(OUT_CH / 16) * (HC1 / 32) * 32 * 8 * 4);
  float*          h1      = (float*)carve((size_t)N * HC1 * 4);
  float*          hout1   = (float*)carve((size_t)N * HC1 * 4);
  unsigned short* helu_bf = (unsigned short*)carve((size_t)N * HC1 * 2);
  float*          h2      = (float*)carve((size_t)N * OUT_CH * 4);
  float*          asrc    = (float*)carve((size_t)N * HEADS * 4);
  float*          adst    = (float*)carve((size_t)N * HEADS * 4);
  unsigned*       nmax    = (unsigned*)carve((size_t)N * HEADS * 4);
  float*          nsum    = (float*)carve((size_t)N * HEADS * 4);
  float*          elog    = (float*)carve((size_t)E * HEADS * 4);

  const int B = 256;
  k_f32_to_bf16<<<(N * IN_CH + B - 1) / B, B, 0, stream>>>(x, x_bf, N * IN_CH);
  k_pack_b_bf16<<<((HC1 / 16) * (IN_CH / 32) * 32 + B - 1) / B, B, 0, stream>>>(W1, w1_pk, HC1, IN_CH);
  k_pack_b_bf16<<<((OUT_CH / 16) * (HC1 / 32) * 32 + B - 1) / B, B, 0, stream>>>(W2, w2_pk, OUT_CH, HC1);

  // ---- Layer 1 ----
  // h1 = x @ W1  (M=50000, N=256, K=128); 4 waves/block, 16 n-tiles => grid.y=4
  k_gemm_bf16_wmma<IN_CH / 32><<<dim3(N / 16, 4), 128, 0, stream>>>(x_bf, w1_pk, h1, N, HC1);

  k_node_att<<<(N * HEADS + B - 1) / B, B, 0, stream>>>(h1, as1, ad1, asrc, adst,
                                                        nmax, nsum, N, HEADS, HID);
  k_fill_bias<<<(N * HC1 + B - 1) / B, B, 0, stream>>>(hout1, b1, N * HC1, HC1);
  k_edge_logits<HEADS><<<(E + B - 1) / B, B, 0, stream>>>(src, dstv, asrc, adst, elog, nmax, E);
  k_edge_exp<HEADS><<<(E + B - 1) / B, B, 0, stream>>>(dstv, elog, nmax, nsum, E);
  {
    long long tot = (long long)E * (HEADS * HID / 4);
    k_edge_scatter<HEADS, HID><<<(unsigned)((tot + B - 1) / B), B, 0, stream>>>(
        src, dstv, h1, elog, nsum, hout1, E);
  }
  k_elu_to_bf16<<<(N * HC1 + B - 1) / B, B, 0, stream>>>(hout1, helu_bf, N * HC1);

  // ---- Layer 2 ----
  // h2 = elu(h) @ W2  (M=50000, N=32, K=256); 2 waves/block cover both n-tiles
  k_gemm_bf16_wmma<HC1 / 32><<<dim3(N / 16, 1), 64, 0, stream>>>(helu_bf, w2_pk, h2, N, OUT_CH);

  k_node_att<<<(N + B - 1) / B, B, 0, stream>>>(h2, as2, ad2, asrc, adst,
                                                nmax, nsum, N, 1, OUT_CH);
  k_fill_bias<<<(N * OUT_CH + B - 1) / B, B, 0, stream>>>(out, b2, N * OUT_CH, OUT_CH);
  k_edge_logits<1><<<(E + B - 1) / B, B, 0, stream>>>(src, dstv, asrc, adst, elog, nmax, E);
  k_edge_exp<1><<<(E + B - 1) / B, B, 0, stream>>>(dstv, elog, nmax, nsum, E);
  {
    long long tot = (long long)E * (OUT_CH / 4);
    k_edge_scatter<1, OUT_CH><<<(unsigned)((tot + B - 1) / B), B, 0, stream>>>(
        src, dstv, h2, elog, nsum, out, E);
  }
  (void)in_sizes; (void)n_in; (void)out_size; (void)ws_size;
}
